// MHSelfAttention_77747497992540
// MI455X (gfx1250) — matrix operand order repaired and checked by reading, not compiled, not run on hardware
//
#include <hip/hip_runtime.h>
#include <hip/hip_bf16.h>
#include <stdint.h>

typedef unsigned short u16;
typedef unsigned int   u32;
typedef unsigned long long u64;
typedef __attribute__((ext_vector_type(16))) __bf16 v16bf;
typedef __attribute__((ext_vector_type(8)))  float  v8f;
typedef __attribute__((ext_vector_type(8)))  u32    v8u;

static constexpr int Bsz = 4, Nseq = 2048, Dm = 1024, Hh = 16, Hd = 64;
static constexpr int Mrows = Bsz * Nseq;   // 8192
static constexpr int Kdim  = Dm;           // 1024

#if defined(__has_builtin)
#if __has_builtin(__builtin_amdgcn_tensor_load_to_lds) && \
    __has_builtin(__builtin_amdgcn_s_wait_tensorcnt)
#define HAVE_TDM 1
#endif
#endif
#ifndef HAVE_TDM
#define HAVE_TDM 0
#endif

// ---------- helpers ----------

__device__ __forceinline__ u16 f2bf(float f) {
  u32 u = __builtin_bit_cast(u32, f);
  u += 0x7FFFu + ((u >> 16) & 1u);   // round-to-nearest-even
  return (u16)(u >> 16);
}
__device__ __forceinline__ u32 pack2bf(float lo, float hi) {
  return (u32)f2bf(lo) | ((u32)f2bf(hi) << 16);
}

__device__ __forceinline__ v8f wmma_bf16(v16bf a, v16bf b, v8f c) {
  // (neg_a, A, neg_b, B, c_mod, C, reuse_a, reuse_b)
  return __builtin_amdgcn_wmma_f32_16x16x32_bf16(false, a, false, b, (short)0, c,
                                                 false, false);
}

// A fragment: 16x32 bf16, source stored row-major [row][k] in LDS (u16 units).
__device__ __forceinline__ v16bf frag_a(const u16* base, int stride) {
  const int lane = threadIdx.x & 31;
  const int row  = lane & 15;
  const int half = lane >> 4;
  v8u t;
#pragma unroll
  for (int i = 0; i < 8; ++i) {
    const int k = 2 * i + 16 * half; // contiguous: k = 16*half + 2i
    t[i] = *(const u32*)(base + row * stride + k);
  }
  return __builtin_bit_cast(v16bf, t);
}

// B fragment: 32x16 (KxN), source stored [n][k] row-major (NT layout).
__device__ __forceinline__ v16bf frag_b_nk(const u16* base, int stride) {
  const int lane = threadIdx.x & 31;
  const int n    = lane & 15;
  const int half = lane >> 4;
  v8u t;
#pragma unroll
  for (int i = 0; i < 8; ++i) {
    const int k = 2 * i + 16 * half;
    t[i] = *(const u32*)(base + n * stride + k);
  }
  return __builtin_bit_cast(v16bf, t);
}

// B fragment: 32x16 (KxN), source stored [k][n] row-major (NN layout, for V).
__device__ __forceinline__ v16bf frag_b_kn(const u16* base, int stride) {
  const int lane = threadIdx.x & 31;
  const int n    = lane & 15;
  const int half = lane >> 4;
  v8u t;
#pragma unroll
  for (int i = 0; i < 8; ++i) {
    const int k  = 2 * i + 16 * half;
    u32 lo = base[k * stride + n];
    u32 hi = base[(k + 1) * stride + n];
    t[i] = lo | (hi << 16);
  }
  return __builtin_bit_cast(v16bf, t);
}

#if HAVE_TDM
typedef __attribute__((ext_vector_type(4))) u32 v4u;
typedef __attribute__((ext_vector_type(8))) int v8i;
typedef __attribute__((ext_vector_type(4))) int v4i;

__device__ __forceinline__ u32 lds_off(const void* p) {
  return (u32)(uintptr_t)(__attribute__((address_space(3))) const void*)p;
}

// TDM 2D load: 64 rows x 64 bf16 (128B) rows, global row stride 128B,
// LDS row stride 144B via pad (interval 32 DWORDs -> code 4, amount 4 DWORDs -> code 3)
__device__ __forceinline__ void tdm_load_2d(u32 lds_byte_off, const void* gaddr) {
  const u64 ga = (u64)(uintptr_t)gaddr;
  v4u g0;
  g0[0] = 1u;                                            // count=1, user mode
  g0[1] = lds_byte_off;                                  // lds_addr
  g0[2] = (u32)(ga & 0xFFFFFFFFu);                       // global_addr[31:0]
  g0[3] = (u32)((ga >> 32) & 0x1FFFFFFu) | (2u << 30);   // addr[56:32], type=2
  v8i g1;
  g1[0] = (int)((1u << 16) | (1u << 20) | (4u << 22) | (3u << 25));
  //            data_size=2B  pad_en      pad_interval   pad_amount
  g1[1] = (int)(64u << 16);        // tensor_dim0 low16 = 64
  g1[2] = (int)(64u << 16);        // tensor_dim0 hi=0, tensor_dim1 low16 = 64
  g1[3] = (int)(64u << 16);        // tensor_dim1 hi=0, tile_dim0 = 64
  g1[4] = 64;                      // tile_dim1 = 64, tile_dim2 = 0
  g1[5] = 64;                      // tensor_dim0_stride low32 = 64 elems
  g1[6] = 0;                       // stride hi, dim1_stride low16
  g1[7] = 0;
  v4i z4 = {0, 0, 0, 0};
  v8i z8 = {0, 0, 0, 0, 0, 0, 0, 0};
  // 6-arg form (amdgpu-toolchain / clang-23): (g0, g1, g2, g3, g_extra, cpol)
  __builtin_amdgcn_tensor_load_to_lds(g0, g1, z4, z4, z8, 0);
}
#endif

// ---------- tiled GEMM:  C[M, NCOLS] = A[M,K] * W[NCOLS,K]^T ----------
// BM=256, BN=128, 8 waves in 4x2 grid, 64x64 wave tile -> 16 wmma per k-step.
// MODE 0: epilogue scatters bf16 QKV to [B,H,N,Hd], scaling Q by 0.125.
// MODE 1: epilogue adds bias and writes f32.

template <int NCOLS, int MODE>
__global__ __launch_bounds__(256) void gemm_kernel(
    const float* __restrict__ A, const float* __restrict__ W,
    const float* __restrict__ bias,
    u16* __restrict__ qo, u16* __restrict__ ko, u16* __restrict__ vo,
    float* __restrict__ outp) {
  constexpr int BM = 256, BN = 128, BK = 32, LS = 34;  // LS padded: conflict-free
  __shared__ u16 sA[BM * LS];
  __shared__ u16 sB[BN * LS];

  const int tid  = threadIdx.x;
  const int lane = tid & 31, wid = tid >> 5;
  const int half = lane >> 4, nl = lane & 15;
  const int wm = (wid >> 1) * 64;   // 4x2 wave grid; wave tile 64(M) x 64(N)
  const int wn = (wid & 1) * 64;
  const int mBase = blockIdx.y * BM;
  const int nBase = blockIdx.x * BN;

  v8f acc[4][4];
#pragma unroll
  for (int i = 0; i < 4; ++i)
#pragma unroll
    for (int j = 0; j < 4; ++j) acc[i][j] = v8f{};

  // A: thread -> one 32-float row slice.  B: 128 rows x 2 segs of 16 floats.
  const float* srcA = A + (size_t)(mBase + tid) * Kdim;
  u32* dstA = (u32*)&sA[tid * LS];
  const int br = tid >> 1, bseg = tid & 1;
  const float* srcB = W + (size_t)(nBase + br) * Kdim + bseg * 16;
  u32* dstB = (u32*)&sB[br * LS + bseg * 16];

  for (int k0 = 0; k0 < Kdim; k0 += BK) {
    __syncthreads();
    {  // A tile: fp32 -> bf16 into LDS
      const float4* s4 = (const float4*)(srcA + k0);
#pragma unroll
      for (int i = 0; i < 8; ++i) {
        float4 f = s4[i];
        dstA[2 * i]     = pack2bf(f.x, f.y);
        dstA[2 * i + 1] = pack2bf(f.z, f.w);
      }
    }
    {  // W tile
      const float4* s4 = (const float4*)(srcB + k0);
#pragma unroll
      for (int i = 0; i < 4; ++i) {
        float4 f = s4[i];
        dstB[2 * i]     = pack2bf(f.x, f.y);
        dstB[2 * i + 1] = pack2bf(f.z, f.w);
      }
    }
    if (k0 + BK < Kdim) {  // prefetch next k-slice -> global_prefetch_b8
      __builtin_prefetch(srcA + k0 + BK);
      __builtin_prefetch(srcB + k0 + BK);
    }
    __syncthreads();

    v16bf af[4], bf[4];
#pragma unroll
    for (int mi = 0; mi < 4; ++mi) af[mi] = frag_a(&sA[(wm + mi * 16) * LS], LS);
#pragma unroll
    for (int ni = 0; ni < 4; ++ni) bf[ni] = frag_b_nk(&sB[(wn + ni * 16) * LS], LS);
#pragma unroll
    for (int mi = 0; mi < 4; ++mi)
#pragma unroll
      for (int ni = 0; ni < 4; ++ni)
        acc[mi][ni] = wmma_bf16(af[mi], bf[ni], acc[mi][ni]);
  }

  // epilogue: C fragment row M = r + 8*half, col N = lane&15
#pragma unroll
  for (int mi = 0; mi < 4; ++mi)
#pragma unroll
    for (int ni = 0; ni < 4; ++ni)
#pragma unroll
      for (int r = 0; r < 8; ++r) {
        const int m = mBase + wm + mi * 16 + r + 8 * half;
        const int e = nBase + wn + ni * 16 + nl;
        float val = acc[mi][ni][r];
        if (MODE == 0) {
          const int which = e >> 10;     // 0=Q 1=K 2=V
          const int d  = e & 1023;
          const int h  = d >> 6;
          const int hd = d & 63;
          const int b  = m >> 11;        // m / Nseq
          const int n  = m & 2047;
          u16* dst = (which == 0) ? qo : (which == 1) ? ko : vo;
          if (which == 0) val *= 0.125f;  // fold 1/sqrt(Hd) into Q
          dst[(((size_t)(b * Hh + h)) * Nseq + n) * Hd + hd] = f2bf(val);
        } else {
          outp[(size_t)m * NCOLS + e] = val + bias[e];
        }
      }
}

// ---------- flash attention per (b,h, 64-row q block) ----------

__global__ __launch_bounds__(128) void attn_kernel(
    const u16* __restrict__ Q, const u16* __restrict__ K,
    const u16* __restrict__ V, float* __restrict__ Oatt) {
  constexpr int LS = 72;  // u16 stride: 144B rows, 16B aligned, conflict-free
  __shared__ u16 sQ[64 * LS];
  __shared__ u16 sK[64 * LS];
  __shared__ u16 sV[64 * LS];
  __shared__ u16 sP[4 * 16 * LS];

  const int tid  = threadIdx.x;
  const int lane = tid & 31, wid = tid >> 5;
  const int half = lane >> 4, nl = lane & 15;
  const int q0 = blockIdx.x * 64;
  const int bh = blockIdx.y;
  const int b = bh / Hh, h = bh % Hh;
  const size_t base = (size_t)bh * Nseq * Hd;
  const u16* Qp = Q + base;
  const u16* Kp = K + base;
  const u16* Vp = V + base;

  const int lr = tid >> 1, seg = tid & 1;  // 64 rows x 2 segs of 32 u16

  {  // load Q tile (already scaled by 0.125 in projection)
    const uint4* src = (const uint4*)(Qp + (size_t)(q0 + lr) * Hd + seg * 32);
    uint4* dst = (uint4*)&sQ[lr * LS + seg * 32];
    dst[0] = src[0]; dst[1] = src[1]; dst[2] = src[2]; dst[3] = src[3];
  }
  __syncthreads();
  const v16bf qa0 = frag_a(&sQ[(wid * 16) * LS], LS);       // k 0..31
  const v16bf qa1 = frag_a(&sQ[(wid * 16) * LS + 32], LS);  // k 32..63

  float mst[8], lst[8];
  v8f o[4];
#pragma unroll
  for (int r = 0; r < 8; ++r) { mst[r] = -__builtin_inff(); lst[r] = 0.f; }
#pragma unroll
  for (int t = 0; t < 4; ++t) o[t] = v8f{};

  const int qrow = q0 + wid * 16 + 8 * half;
  const int nkb = blockIdx.x + 1;  // causal: only key blocks <= q block

#if HAVE_TDM
  const u32 offK = lds_off(&sK[0]);
  const u32 offV = lds_off(&sV[0]);
#endif

  for (int kb = 0; kb < nkb; ++kb) {
    __syncthreads();  // previous iteration done reading sK/sV
#if HAVE_TDM
    if (wid == 0) {  // one wave drives the Tensor Data Mover for both tiles
      tdm_load_2d(offK, Kp + (size_t)(kb * 64) * Hd);
      tdm_load_2d(offV, Vp + (size_t)(kb * 64) * Hd);
      __builtin_amdgcn_s_wait_tensorcnt(0);
    }
#else
    {  // stage K, V blocks with vector loads
      const uint4* sk = (const uint4*)(Kp + (size_t)(kb * 64 + lr) * Hd + seg * 32);
      uint4* dk = (uint4*)&sK[lr * LS + seg * 32];
      dk[0] = sk[0]; dk[1] = sk[1]; dk[2] = sk[2]; dk[3] = sk[3];
      const uint4* sv = (const uint4*)(Vp + (size_t)(kb * 64 + lr) * Hd + seg * 32);
      uint4* dv = (uint4*)&sV[lr * LS + seg * 32];
      dv[0] = sv[0]; dv[1] = sv[1]; dv[2] = sv[2]; dv[3] = sv[3];
    }
#endif
    __syncthreads();

    // S = Q K^T : wave's 16 q-rows x 64 keys, f32 accum
    v8f s[4];
#pragma unroll
    for (int nt = 0; nt < 4; ++nt) {
      v16bf kb0 = frag_b_nk(&sK[(nt * 16) * LS], LS);
      v16bf kb1 = frag_b_nk(&sK[(nt * 16) * LS + 32], LS);
      v8f a = v8f{};
      a = wmma_bf16(qa0, kb0, a);
      a = wmma_bf16(qa1, kb1, a);
      s[nt] = a;
    }

    // causal mask + running max
    float mnew[8];
#pragma unroll
    for (int r = 0; r < 8; ++r) mnew[r] = mst[r];
#pragma unroll
    for (int nt = 0; nt < 4; ++nt) {
      const int kg = kb * 64 + nt * 16 + nl;
#pragma unroll
      for (int r = 0; r < 8; ++r) {
        float sv = s[nt][r];
        sv = (kg > qrow + r) ? -__builtin_inff() : sv;
        s[nt][r] = sv;
        mnew[r] = fmaxf(mnew[r], sv);
      }
    }
#pragma unroll
    for (int r = 0; r < 8; ++r) {  // reduce across the 16 lanes of this half
      mnew[r] = fmaxf(mnew[r], __shfl_xor(mnew[r], 8));
      mnew[r] = fmaxf(mnew[r], __shfl_xor(mnew[r], 4));
      mnew[r] = fmaxf(mnew[r], __shfl_xor(mnew[r], 2));
      mnew[r] = fmaxf(mnew[r], __shfl_xor(mnew[r], 1));
    }

    // P = exp(S - mnew); write bf16 P to per-wave LDS; accumulate row sums
    float rs[8];
#pragma unroll
    for (int r = 0; r < 8; ++r) rs[r] = 0.f;
    u16* pbase = &sP[wid * 16 * LS];
#pragma unroll
    for (int nt = 0; nt < 4; ++nt)
#pragma unroll
      for (int r = 0; r < 8; ++r) {
        float p = __expf(s[nt][r] - mnew[r]);
        rs[r] += p;
        pbase[(r + 8 * half) * LS + nt * 16 + nl] = f2bf(p);
      }
#pragma unroll
    for (int r = 0; r < 8; ++r) {
      rs[r] += __shfl_xor(rs[r], 8);
      rs[r] += __shfl_xor(rs[r], 4);
      rs[r] += __shfl_xor(rs[r], 2);
      rs[r] += __shfl_xor(rs[r], 1);
      const float cf = __expf(mst[r] - mnew[r]);
      lst[r] = lst[r] * cf + rs[r];
      mst[r] = mnew[r];
#pragma unroll
      for (int t = 0; t < 4; ++t) o[t][r] *= cf;
    }

    // ensure our P stores are visible to our own fragment loads (wave-local)
    asm volatile("s_wait_dscnt 0" ::: "memory");

    // O += P * V
    const v16bf pa0 = frag_a(pbase, LS);       // keys 0..31
    const v16bf pa1 = frag_a(pbase + 32, LS);  // keys 32..63
#pragma unroll
    for (int t = 0; t < 4; ++t) {
      v16bf vb0 = frag_b_kn(&sV[t * 16], LS);
      v16bf vb1 = frag_b_kn(&sV[32 * LS + t * 16], LS);
      o[t] = wmma_bf16(pa0, vb0, o[t]);
      o[t] = wmma_bf16(pa1, vb1, o[t]);
    }
  }

  // normalize and write f32 attn output as [B, N, D] (D = h*64 + hd)
#pragma unroll
  for (int r = 0; r < 8; ++r) {
    const float inv = 1.0f / lst[r];
    const int n = q0 + wid * 16 + r + 8 * half;
    float* dst = Oatt + ((size_t)(b * Nseq + n)) * Dm + h * Hd + nl;
#pragma unroll
    for (int t = 0; t < 4; ++t) dst[t * 16] = o[t][r] * inv;
  }
}

// ---------- launch ----------

extern "C" void kernel_launch(void* const* d_in, const int* in_sizes, int n_in,
                              void* d_out, int out_size, void* d_ws, size_t ws_size,
                              hipStream_t stream) {
  const float* x     = (const float*)d_in[0];  // [4,2048,1024]
  const float* w_in  = (const float*)d_in[1];  // [3072,1024]
  const float* w_out = (const float*)d_in[2];  // [1024,1024]
  const float* b_out = (const float*)d_in[3];  // [1024]
  float* out = (float*)d_out;                  // [4,2048,1024]

  const size_t QKV_ELEMS = (size_t)Bsz * Hh * Nseq * Hd;  // 8388608
  u16* q = (u16*)d_ws;
  u16* k = q + QKV_ELEMS;
  u16* v = k + QKV_ELEMS;
  float* attn = (float*)(v + QKV_ELEMS);  // 32 MB f32, 16B aligned

  // 1) QKV projection: [8192,1024] x [3072,1024]^T -> scattered bf16 Q/K/V
  gemm_kernel<3 * Dm, 0><<<dim3((3 * Dm) / 128, Mrows / 256), 256, 0, stream>>>(
      x, w_in, nullptr, q, k, v, nullptr);

  // 2) causal flash attention per (b*h, q-block)
  attn_kernel<<<dim3(Nseq / 64, Bsz * Hh), 128, 0, stream>>>(q, k, v, attn);

  // 3) output projection + bias: [8192,1024] x [1024,1024]^T + b
  gemm_kernel<Dm, 1><<<dim3(Dm / 128, Mrows / 256), 256, 0, stream>>>(
      attn, w_out, b_out, nullptr, nullptr, nullptr, out);
}
